// Head_18537078849687
// MI455X (gfx1250) — matrix operand order repaired and checked by reading, not compiled, not run on hardware
//
#include <hip/hip_runtime.h>

// ---------------------------------------------------------------------------
// Causal single-head attention, B=32768, T=32, C=H=64, fp32 in/out.
// One wave32 per batch; all GEMMs via v_wmma_f32_16x16x32_f16 (fp32 acc).
// The q/k projection GEMMs are computed in the orientation whose *transposed* D store
// (lane = column -> 8 contiguous rows per lane) is a single ds_store_b128,
// landing in exactly the layout the next GEMM's A/B fragment reads need.
// ---------------------------------------------------------------------------

typedef _Float16 h16 __attribute__((ext_vector_type(16)));
typedef _Float16 h8  __attribute__((ext_vector_type(8)));
typedef float    f8  __attribute__((ext_vector_type(8)));
typedef float    f4  __attribute__((ext_vector_type(4)));

constexpr int TT = 32;   // sequence length
constexpr int CC = 64;   // embed = head size
constexpr int WAVES_PER_BLOCK = 4;
constexpr int NBLOCKS = 1024;
constexpr int WAVE_SCRATCH = 19456;           // 19KB per-wave LDS scratch
constexpr int WEIGHT_LDS   = 3 * 64 * 64 * 2; // 24KB f16 weights

#define WMMA_F16(A, B, Cacc) \
  __builtin_amdgcn_wmma_f32_16x16x32_f16(false, (A), false, (B), (short)0, (Cacc), false, false)

// DS ops from one wave are in-order; wait for LDS writes to land before
// cross-lane reads (compiler cannot see cross-lane dependencies).
#define LDS_FENCE() asm volatile("s_wait_dscnt 0" ::: "memory")

__device__ __forceinline__ h16 cat8(h8 lo, h8 hi) {
  return __builtin_shufflevector(lo, hi, 0,1,2,3,4,5,6,7,8,9,10,11,12,13,14,15);
}

// A fragment (16x32 f16) from row-major f16 LDS, row stride `stride` halves.
// Lane m<16: row M=m, K = kb+0..15 contiguous; lane m+16: K = kb+16..31.
__device__ __forceinline__ h16 ldsA(const _Float16* buf, int stride,
                                    int rowBase, int kb, int lane) {
  int m = lane & 15, hf = lane >> 4;
  const _Float16* p = buf + (rowBase + m) * stride + kb + hf * 16;
  h8 lo = *(const h8*)(p);
  h8 hi = *(const h8*)(p + 8);
  return cat8(lo, hi);
}

// B fragment (32x16 f16): lane n<16 holds column N=n, K=kb+0..15 contiguous;
// lane n+16 holds K=kb+16..31.  Source buffer has one row per output column.
__device__ __forceinline__ h16 ldsB(const _Float16* buf, int stride,
                                    int colBase, int kb, int lane) {
  int n = lane & 15, hf = lane >> 4;
  return *(const h16*)(buf + (colBase + n) * stride + kb + hf * 16);
}

// B fragment of x^T straight from global fp32 (column t = row t of x).
__device__ __forceinline__ h16 gblB(const float* xb, int colBase, int kb, int lane) {
  int n = lane & 15, hf = lane >> 4;
  const float* p = xb + (colBase + n) * CC + kb + hf * 16;
  f4 c0 = *(const f4*)(p);
  f4 c1 = *(const f4*)(p + 4);
  f4 c2 = *(const f4*)(p + 8);
  f4 c3 = *(const f4*)(p + 12);
  h16 r;
#pragma unroll
  for (int i = 0; i < 4; ++i) {
    r[i]      = (_Float16)c0[i];
    r[4 + i]  = (_Float16)c1[i];
    r[8 + i]  = (_Float16)c2[i];
    r[12 + i] = (_Float16)c3[i];
  }
  return r;
}

__device__ __forceinline__ h8 packD(f8 acc) {
  h8 r;
#pragma unroll
  for (int i = 0; i < 8; ++i) r[i] = (_Float16)acc[i];
  return r;
}

__global__ __launch_bounds__(WAVES_PER_BLOCK * 32)
void head_attn_wmma(const float* __restrict__ x,
                    const float* __restrict__ Wq,
                    const float* __restrict__ Wk,
                    const float* __restrict__ Wv,
                    float* __restrict__ out, int nB) {
  extern __shared__ __align__(128) char smem[];
  _Float16* WqS = (_Float16*)smem;              // [64][64] f16, row-major [h][c]
  _Float16* WkS = WqS + 4096;
  _Float16* WvS = WkS + 4096;

  const int tid = threadIdx.x;
  for (int i = tid; i < 4096; i += blockDim.x) {
    WqS[i] = (_Float16)Wq[i];
    WkS[i] = (_Float16)Wk[i];
    WvS[i] = (_Float16)Wv[i];
  }
  __syncthreads();

  const int wave = tid >> 5, lane = tid & 31;
  const int hf = lane >> 4, mn = lane & 15;

  char* wbase = smem + WEIGHT_LDS + wave * WAVE_SCRATCH;
  _Float16* qS = (_Float16*)wbase;              // [32][64] f16  q row-major [t][h]
  _Float16* kS = qS + 2048;                     // [32][64] f16  k row-major [t][h]
  _Float16* vT = kS + 2048;                     // [64][32] f16  v^T row-major [h][t]
  float*    wF = (float*)(wbase + 12288);       // [32][32] f32  logits, [s][t] layout
  _Float16* wH = (_Float16*)(wbase + 16384);    // [32][32] f16  softmax, [t][s] layout
  float*    oS = (float*)wbase;                 // [32][64] f32  out stage (reuses qS/kS)

  // 32768 / (1024 blocks * 4 waves) = 8 full iterations; wave-uniform loop,
  // EXEC stays all-ones for every WMMA.
  for (int b = blockIdx.x * WAVES_PER_BLOCK + wave; b < nB;
       b += NBLOCKS * WAVES_PER_BLOCK) {
    const float* xb = x + (size_t)b * (TT * CC);

    // ---- x^T B-fragments (in VGPRs, shared by all three projections) ----
    h16 xB[2][2];
#pragma unroll
    for (int nt = 0; nt < 2; ++nt)
#pragma unroll
      for (int kk = 0; kk < 2; ++kk)
        xB[nt][kk] = gblB(xb, nt * 16, kk * 32, lane);

    // ---- projections: p^T = W * x^T  (M = h-tiles 4, N = t-tiles 2, K = c) ----
    // q,k: D = p^T[h][t]; transposed store -> row-major [t][h], 1x ds_store_b128.
    // v  : needs v^T row-major [h][t] = D row-major -> scalar stores.
#pragma unroll
    for (int w = 0; w < 3; ++w) {
      const _Float16* W = (w == 0) ? WqS : (w == 1) ? WkS : WvS;
#pragma unroll
      for (int mt = 0; mt < 4; ++mt) {
        h16 a0 = ldsA(W, 64, mt * 16, 0, lane);
        h16 a1 = ldsA(W, 64, mt * 16, 32, lane);
#pragma unroll
        for (int nt = 0; nt < 2; ++nt) {
          f8 acc = {};
          acc = WMMA_F16(a0, xB[nt][0], acc);
          acc = WMMA_F16(a1, xB[nt][1], acc);
          if (w < 2) {
            _Float16* dst = (w == 0) ? qS : kS;
            *(h8*)(dst + (nt * 16 + mn) * 64 + mt * 16 + hf * 8) = packD(acc);
          } else {
#pragma unroll
            for (int r = 0; r < 8; ++r)
              vT[(mt * 16 + hf * 8 + r) * 32 + nt * 16 + mn] = (_Float16)acc[r];
          }
        }
      }
    }
    LDS_FENCE();

    // ---- wei = q k^T : A = q rows, B = k rows.  Tile (t0,s1) fully masked,
    //      skipped.  D = wei[t][s]; transposed store -> wF[s][t] (f4 x2). ----
#pragma unroll
    for (int mt = 0; mt < 2; ++mt)
#pragma unroll
      for (int st = 0; st < 2; ++st) {
        if (mt == 0 && st == 1) continue;     // s > t everywhere: masked tile
        f8 acc = {};
#pragma unroll
        for (int kk = 0; kk < 2; ++kk)
          acc = WMMA_F16(ldsA(qS, 64, mt * 16, kk * 32, lane),
                         ldsB(kS, 64, st * 16, kk * 32, lane), acc);
        float* p = wF + (st * 16 + mn) * 32 + mt * 16 + hf * 8;
        *(f4*)(p)     = __builtin_shufflevector(acc, acc, 0, 1, 2, 3);
        *(f4*)(p + 4) = __builtin_shufflevector(acc, acc, 4, 5, 6, 7);
      }
    LDS_FENCE();

    // ---- causal softmax: lane t owns row t; wF is [s][t] so the column
    //      reads are stride-1 across lanes (bank-conflict-free). ----
    {
      const int t = lane;
      float row[32];
#pragma unroll
      for (int s = 0; s < 32; ++s) row[s] = wF[s * 32 + t];
      float mx = -1e30f;
#pragma unroll
      for (int s = 0; s < 32; ++s)
        if (s <= t) mx = fmaxf(mx, row[s] * 0.125f);   // scale = C^-0.5
      float sum = 0.f;
#pragma unroll
      for (int s = 0; s < 32; ++s) {
        float e = (s <= t) ? __expf(row[s] * 0.125f - mx) : 0.f;
        row[s] = e;
        sum += e;
      }
      float inv = 1.f / sum;
#pragma unroll
      for (int j = 0; j < 4; ++j) {          // packed h8 stores, row-major [t][s]
        h8 pk;
#pragma unroll
        for (int i = 0; i < 8; ++i) pk[i] = (_Float16)(row[j * 8 + i] * inv);
        *(h8*)(wH + t * 32 + j * 8) = pk;
      }
    }
    LDS_FENCE();

    // ---- out^T = v^T wei^T : A = v^T rows (K=s), B = wei rows (wH).
    //      D = out^T[h][t]; transposed store -> oS row-major [t][h]. ----
    h16 bw[2];
#pragma unroll
    for (int nt = 0; nt < 2; ++nt) bw[nt] = ldsB(wH, 32, nt * 16, 0, lane);
#pragma unroll
    for (int mt = 0; mt < 4; ++mt) {
      h16 a = ldsA(vT, 32, mt * 16, 0, lane);
#pragma unroll
      for (int nt = 0; nt < 2; ++nt) {
        f8 acc = {};
        acc = WMMA_F16(a, bw[nt], acc);
        float* p = oS + (nt * 16 + mn) * 64 + mt * 16 + hf * 8;
        *(f4*)(p)     = __builtin_shufflevector(acc, acc, 0, 1, 2, 3);
        *(f4*)(p + 4) = __builtin_shufflevector(acc, acc, 4, 5, 6, 7);
      }
    }
    LDS_FENCE();

    // ---- coalesced fp32 output: 2048 floats via b128 stores ----
    float* ob = out + (size_t)b * (TT * CC);
#pragma unroll
    for (int i = 0; i < 16; ++i) {
      int idx = (i * 32 + lane) * 4;
      *(f4*)(ob + idx) = *(const f4*)(oS + idx);
    }
  }
}

extern "C" void kernel_launch(void* const* d_in, const int* in_sizes, int n_in,
                              void* d_out, int out_size, void* d_ws, size_t ws_size,
                              hipStream_t stream) {
  const float* x  = (const float*)d_in[0];
  const float* Wq = (const float*)d_in[1];
  const float* Wk = (const float*)d_in[2];
  const float* Wv = (const float*)d_in[3];
  float* out = (float*)d_out;
  int nB = in_sizes[0] / (TT * CC);   // 32768

  size_t shmem = WEIGHT_LDS + WAVES_PER_BLOCK * WAVE_SCRATCH;  // 100 KB
  hipLaunchKernelGGL(head_attn_wmma, dim3(NBLOCKS), dim3(WAVES_PER_BLOCK * 32),
                     shmem, stream, x, Wq, Wk, Wv, out, nB);
}
